// PKM_32796370272951
// MI455X (gfx1250) — matrix operand order repaired and checked by reading, not compiled, not run on hardware
//
#include <hip/hip_runtime.h>
#include <hip/hip_bf16.h>

// ---------------------------------------------------------------------------
// PKM (product-key memory) attention for MI455X / gfx1250, wave32 + WMMA f16,
// async global->LDS staging of shared A tiles (ASYNCcnt path).
// Shapes fixed: t=1024, b=2, dim=1024, h=4, n=512, dh2=128, TOPK=32.
// ---------------------------------------------------------------------------

#define TB        2048        // t*b tokens
#define DIM       1024        // model dim == dim_query
#define HEADS     4
#define NKEYS     512
#define DH2       128
#define TOPK      32
#define HK        128         // HEADS*TOPK

typedef __attribute__((ext_vector_type(16))) _Float16 v16h;
typedef __attribute__((ext_vector_type(8)))  _Float16 v8h;
typedef __attribute__((ext_vector_type(4)))  _Float16 v4h;
typedef __attribute__((ext_vector_type(8)))  float    v8f;
typedef __attribute__((ext_vector_type(4)))  int      v4i;

union V16U { v16h v; v8h h[2]; };

// ---------------------------------------------------------------------------
// Async global -> LDS staging (gfx1250 GLOBAL_LOAD_ASYNC_TO_LDS_B128).
// Builtin confirmed present; parameter type is v4i* in AS(1)/AS(3).
// ---------------------------------------------------------------------------
#if __has_builtin(__builtin_amdgcn_global_load_async_to_lds_b128)
#define HAVE_ASYNC_LDS 1
#else
#define HAVE_ASYNC_LDS 0
#endif

typedef __attribute__((address_space(1))) v4i* as1_v4i;
typedef __attribute__((address_space(3))) v4i* as3_v4i;

// copy 32 bytes (16 halves) global -> LDS
__device__ __forceinline__ void stage32B(_Float16* l, const _Float16* g) {
#if HAVE_ASYNC_LDS
    as1_v4i gp = (as1_v4i)g;     // C-style cast: const_cast + addrspacecast
    as3_v4i lp = (as3_v4i)l;
    __builtin_amdgcn_global_load_async_to_lds_b128(gp,     lp,     0, 0);
    __builtin_amdgcn_global_load_async_to_lds_b128(gp + 1, lp + 1, 0, 0);
#else
    *(v8h*)l       = *(const v8h*)g;
    *(v8h*)(l + 8) = *(const v8h*)(g + 8);
#endif
}

__device__ __forceinline__ void wait_async() {
#if HAVE_ASYNC_LDS
#if __has_builtin(__builtin_amdgcn_s_wait_asynccnt)
    __builtin_amdgcn_s_wait_asynccnt(0);
#else
    asm volatile("s_wait_asynccnt 0x0" ::: "memory");
#endif
#endif
}

// ---------------------------------------------------------------------------
// WMMA fragments
// ---------------------------------------------------------------------------
// A-fragment (16x32 f16, ISA 7.12.2): lane row m=L&15, hi=L>>4.
// Per-lane K: hi*16 + 0..15 contiguous.  p includes the hi*16 term.
// Works for both global and LDS pointers (inlined -> ds_load_b128 for LDS).
__device__ __forceinline__ v16h load_afrag(const _Float16* p) {
    V16U u;
    u.h[0] = *(const v8h*)(p);
    u.h[1] = *(const v8h*)(p + 8);
    return u.v;
}

// B-fragment (32x16 f16): lane col n=L&15; K = hi*16 + 0..15 contiguous.
__device__ __forceinline__ v16h load_bfrag(const _Float16* p) {
    V16U u;
    u.h[0] = *(const v8h*)(p);
    u.h[1] = *(const v8h*)(p + 8);
    return u.v;
}

// ---------------------------------------------------------------------------
// Conversions
// ---------------------------------------------------------------------------
__global__ void __launch_bounds__(256)
k_f32_to_f16(const float* __restrict__ in, _Float16* __restrict__ out) {
    int i = blockIdx.x * 256 + threadIdx.x;       // element-quad index
    float4 v = ((const float4*)in)[i];
    v4h o = { (_Float16)v.x, (_Float16)v.y, (_Float16)v.z, (_Float16)v.w };
    *(v4h*)(out + (size_t)i * 4) = o;
}

// Wq[e][f] (1024x1024) -> WqT_h[f][e]
__global__ void __launch_bounds__(256)
k_wq_transpose_f16(const float* __restrict__ wq, _Float16* __restrict__ wqT) {
    int idx = blockIdx.x * 256 + threadIdx.x;     // over 1M
    int e = idx >> 10, f = idx & 1023;
    wqT[f * DIM + e] = (_Float16)wq[idx];
}

// ---------------------------------------------------------------------------
// Kernel 1: q_raw = x * Wq   (2048x1024, K=1024), f16 WMMA, f32 accum.
// Block = 4 waves sharing one 16-row A strip; A staged into LDS by async
// copies, double-buffered in 128-wide K stages (4 WMMA k-steps per stage).
// Each wave owns a 16x64 output tile (4 accumulators).
// ---------------------------------------------------------------------------
__global__ void __launch_bounds__(128)
k_qproj(const _Float16* __restrict__ xb, const _Float16* __restrict__ wqT,
        float* __restrict__ qraw) {
    __shared__ _Float16 Atile[2][16][128];        // 2 x 4 KB
    const int tid  = threadIdx.x;
    const int lane = tid & 31;
    const int wave = tid >> 5;
    const int rt   = blockIdx.x >> 2;             // row tile    0..127
    const int cq   = blockIdx.x & 3;              // col quad    0..3
    const int cg   = cq * 4 + wave;               // col group   0..15 (64 cols)
    const int ln   = lane & 15;
    const int hi   = lane >> 4;
    const int col0 = cg * 64;

    // staging geometry: 128 threads x 32B = 4 KB per stage
    const int crow = tid >> 3;                    // 0..15
    const int cseg = tid & 7;                     // 0..7 (16-half segments)
    const _Float16* gsrc = xb + (size_t)(rt * 16 + crow) * DIM + cseg * 16;

    v8f acc[4] = {};
    stage32B(&Atile[0][crow][cseg * 16], gsrc);   // prologue: stage 0

    for (int s = 0; s < 8; ++s) {                 // 8 stages x 128 K
        wait_async();
        __syncthreads();                          // stage s visible to all
        if (s + 1 < 8)
            stage32B(&Atile[(s + 1) & 1][crow][cseg * 16],
                     gsrc + (size_t)(s + 1) * 128);
        const _Float16* At = &Atile[s & 1][0][0];
#pragma unroll
        for (int k4 = 0; k4 < 4; ++k4) {          // 4 WMMA k-steps of 32
            const int kk = k4 * 32;
            v16h a = load_afrag(At + ln * 128 + kk + hi * 16);
            v16h b[4];
#pragma unroll
            for (int t4 = 0; t4 < 4; ++t4)
                b[t4] = load_bfrag(wqT + (size_t)(col0 + t4 * 16 + ln) * DIM +
                                   s * 128 + kk + hi * 16);
#pragma unroll
            for (int t4 = 0; t4 < 4; ++t4)
                acc[t4] = __builtin_amdgcn_wmma_f32_16x16x32_f16(
                    false, a, false, b[t4], (short)0, acc[t4], false, false);
        }
    }
#pragma unroll
    for (int t4 = 0; t4 < 4; ++t4) {
#pragma unroll
        for (int c = 0; c < 8; ++c) {
            int r = rt * 16 + c + hi * 8;
            int n = col0 + t4 * 16 + ln;
            qraw[(size_t)r * DIM + n] = acc[t4][c];
        }
    }
}

// ---------------------------------------------------------------------------
// Kernel 2: LayerNorm(1024) per token, emit f16 q for the dots GEMM.
// ---------------------------------------------------------------------------
__global__ void __launch_bounds__(256)
k_lnorm(const float* __restrict__ qraw, const float* __restrict__ gamma,
        const float* __restrict__ beta, _Float16* __restrict__ qh) {
    __shared__ float s_sum[8], s_sq[8];
    const int row = blockIdx.x;
    const int tid = threadIdx.x;
    float4 v = ((const float4*)(qraw + (size_t)row * DIM))[tid];
    float s  = v.x + v.y + v.z + v.w;
    float sq = v.x * v.x + v.y * v.y + v.z * v.z + v.w * v.w;
#pragma unroll
    for (int off = 16; off > 0; off >>= 1) {
        s  += __shfl_xor(s,  off, 32);
        sq += __shfl_xor(sq, off, 32);
    }
    if ((tid & 31) == 0) { s_sum[tid >> 5] = s; s_sq[tid >> 5] = sq; }
    __syncthreads();
    float tot = 0.f, totq = 0.f;
#pragma unroll
    for (int i = 0; i < 8; ++i) { tot += s_sum[i]; totq += s_sq[i]; }
    float mean = tot * (1.0f / DIM);
    float var  = totq * (1.0f / DIM) - mean * mean;
    float rstd = rsqrtf(var + 1e-5f);
    float4 g  = ((const float4*)gamma)[tid];
    float4 bb = ((const float4*)beta)[tid];
    v4h o = { (_Float16)((v.x - mean) * rstd * g.x + bb.x),
              (_Float16)((v.y - mean) * rstd * g.y + bb.y),
              (_Float16)((v.z - mean) * rstd * g.z + bb.z),
              (_Float16)((v.w - mean) * rstd * g.w + bb.w) };
    *(v4h*)(qh + (size_t)row * DIM + tid * 4) = o;
}

// ---------------------------------------------------------------------------
// Kernel 3: dots[tb,h,p,n] = q_p[tb,h,:] . keys[h,n,p,:]  (K=128), WMMA.
// Block = 4 waves sharing one (rt,h,p) A tile (16x128 = 4 KB) staged async
// into LDS once; each wave computes a 16x64 column slice of the 512 keys.
// ---------------------------------------------------------------------------
__global__ void __launch_bounds__(128)
k_dots(const _Float16* __restrict__ qh, const _Float16* __restrict__ keysh,
       float* __restrict__ dots) {
    __shared__ _Float16 Atile[16][128];           // 4 KB
    const int tid  = threadIdx.x;
    const int lane = tid & 31;
    const int wave = tid >> 5;
    const int bid  = blockIdx.x;                  // 0..2047
    const int rt   = bid >> 4;                    // 0..127
    const int hp   = (bid >> 1) & 7;              // h*2+p
    const int cg   = (bid & 1) * 4 + wave;        // 0..7 (64-col groups)
    const int h = hp >> 1, p = hp & 1;
    const int ln = lane & 15, hi = lane >> 4;
    const int koff = p * 512 + h * 128;           // q column window

    // stage the shared A tile: 128 threads x 32B = 4 KB
    const int crow = tid >> 3;
    const int cseg = tid & 7;
    stage32B(&Atile[crow][cseg * 16],
             qh + (size_t)(rt * 16 + crow) * DIM + koff + cseg * 16);
    wait_async();
    __syncthreads();

    v8f acc[4] = {};
#pragma unroll
    for (int k4 = 0; k4 < 4; ++k4) {
        const int kk = k4 * 32;
        v16h a = load_afrag(&Atile[ln][kk + hi * 16]);
        v16h b[4];
#pragma unroll
        for (int t4 = 0; t4 < 4; ++t4) {
            int n = cg * 64 + t4 * 16 + ln;
            b[t4] = load_bfrag(keysh + (size_t)((h * NKEYS + n) * 2 + p) * DH2 +
                               kk + hi * 16);
        }
#pragma unroll
        for (int t4 = 0; t4 < 4; ++t4)
            acc[t4] = __builtin_amdgcn_wmma_f32_16x16x32_f16(
                false, a, false, b[t4], (short)0, acc[t4], false, false);
    }
#pragma unroll
    for (int t4 = 0; t4 < 4; ++t4) {
#pragma unroll
        for (int c = 0; c < 8; ++c) {
            int r = rt * 16 + c + hi * 8;
            int n = cg * 64 + t4 * 16 + ln;
            dots[((size_t)(r * HEADS + h) * 2 + p) * NKEYS + n] = acc[t4][c];
        }
    }
}

// ---------------------------------------------------------------------------
// Kernel 4: top-32 of 512 per (tb,h,p) row.  One wave per row, values in
// registers (16/lane), 32 rounds of masked argmax + shfl_xor tree reduce.
// ---------------------------------------------------------------------------
__global__ void __launch_bounds__(128)
k_top32_512(const float* __restrict__ dots, float* __restrict__ sc,
            int* __restrict__ ix) {
    const int lane = threadIdx.x & 31;
    const int wave = threadIdx.x >> 5;
    const int row  = blockIdx.x * 4 + wave;       // 0..16383
    const float* rp = dots + (size_t)row * NKEYS;
    float v[16];
#pragma unroll
    for (int i = 0; i < 16; ++i) v[i] = rp[i * 32 + lane];
    unsigned taken = 0;
    float myv = 0.f; int myi = 0;
    for (int s = 0; s < TOPK; ++s) {
        float best = -3.4e38f; int bn = 0x7fffffff;
#pragma unroll
        for (int i = 0; i < 16; ++i)
            if (!(taken & (1u << i)) && v[i] > best) { best = v[i]; bn = i * 32 + lane; }
#pragma unroll
        for (int off = 16; off > 0; off >>= 1) {
            float ov = __shfl_xor(best, off, 32);
            int   on = __shfl_xor(bn,   off, 32);
            if (ov > best || (ov == best && on < bn)) { best = ov; bn = on; }
        }
        if ((bn & 31) == lane) taken |= 1u << (bn >> 5);
        if (lane == s) { myv = best; myi = bn; }
    }
    sc[(size_t)row * TOPK + lane] = myv;
    ix[(size_t)row * TOPK + lane] = myi;
}

// ---------------------------------------------------------------------------
// Kernel 5: per (tb,h): combine s1[k1]+s2[k2] (1024 combos), top-32, softmax,
// decode value indices i1*512+i2.  One wave per (tb,h).
// ---------------------------------------------------------------------------
__global__ void __launch_bounds__(128)
k_combine(const float* __restrict__ sc, const int* __restrict__ ix,
          float* __restrict__ attn, int* __restrict__ vidx) {
    const int lane = threadIdx.x & 31;
    const int wave = threadIdx.x >> 5;
    const int th   = blockIdx.x * 4 + wave;       // tb*HEADS + h, 0..8191
    const int b1 = (th * 2 + 0) * TOPK;
    const int b2 = (th * 2 + 1) * TOPK;
    float s1 = sc[b1 + lane]; int i1 = ix[b1 + lane];
    float s2 = sc[b2 + lane]; int i2 = ix[b2 + lane];

    float c[32];                                  // combos for k1 = lane
#pragma unroll
    for (int j = 0; j < 32; ++j) c[j] = s1 + __shfl(s2, j, 32);

    unsigned taken = 0;
    float selv = 0.f; int selc = 0;
    for (int s = 0; s < TOPK; ++s) {
        float best = -3.4e38f; int bc = 0x7fffffff;
#pragma unroll
        for (int j = 0; j < 32; ++j)
            if (!(taken & (1u << j)) && c[j] > best) { best = c[j]; bc = lane * 32 + j; }
#pragma unroll
        for (int off = 16; off > 0; off >>= 1) {
            float ov = __shfl_xor(best, off, 32);
            int   oc = __shfl_xor(bc,   off, 32);
            if (ov > best || (ov == best && oc < bc)) { best = ov; bc = oc; }
        }
        if ((bc >> 5) == lane) taken |= 1u << (bc & 31);
        if (lane == s) { selv = best; selc = bc; }
    }
    float m = selv;
#pragma unroll
    for (int off = 16; off > 0; off >>= 1) m = fmaxf(m, __shfl_xor(m, off, 32));
    float e = __expf(selv - m);
    float sum = e;
#pragma unroll
    for (int off = 16; off > 0; off >>= 1) sum += __shfl_xor(sum, off, 32);
    float a = e / sum;
    int k1 = selc >> 5, k2 = selc & 31;
    int vi = __shfl(i1, k1, 32) * NKEYS + __shfl(i2, k2, 32);
    attn[(size_t)th * TOPK + lane] = a;
    vidx[(size_t)th * TOPK + lane] = vi;
}

// ---------------------------------------------------------------------------
// Kernel 6: out[tb,:] = sum_j attn[tb,j] * values[vidx[tb,j], :]
// HBM-bound gather stage (~1.07 GB): block per token, coalesced 4 KB rows,
// prefetch next row (global_prefetch_b8).
// ---------------------------------------------------------------------------
__global__ void __launch_bounds__(256)
k_output(const float* __restrict__ attn, const int* __restrict__ vidx,
         const float* __restrict__ values, float* __restrict__ out) {
    __shared__ float s_a[HK];
    __shared__ int   s_i[HK];
    const int tb  = blockIdx.x;
    const int tid = threadIdx.x;
    if (tid < HK) {
        s_a[tid] = attn[(size_t)tb * HK + tid];
        s_i[tid] = vidx[(size_t)tb * HK + tid];
    }
    __syncthreads();
    const int c0 = tid * 4;
    float4 acc = {0.f, 0.f, 0.f, 0.f};
    for (int j = 0; j < HK; ++j) {
        const float* vp = values + (size_t)s_i[j] * DIM + c0;
        if (j + 1 < HK)
            __builtin_prefetch(values + (size_t)s_i[j + 1] * DIM + c0, 0, 1);
        float4 v = *(const float4*)vp;
        float  w = s_a[j];
        acc.x += w * v.x; acc.y += w * v.y; acc.z += w * v.z; acc.w += w * v.w;
    }
    *(float4*)(out + (size_t)tb * DIM + c0) = acc;
}

// ---------------------------------------------------------------------------
// Launch
// ---------------------------------------------------------------------------
extern "C" void kernel_launch(void* const* d_in, const int* in_sizes, int n_in,
                              void* d_out, int out_size, void* d_ws, size_t ws_size,
                              hipStream_t stream) {
    const float* x      = (const float*)d_in[0];  // (t,b,dim)
    const float* Wq     = (const float*)d_in[1];  // (dim, dim_query)
    const float* gamma  = (const float*)d_in[2];  // (1024)
    const float* beta   = (const float*)d_in[3];  // (1024)
    const float* keys   = (const float*)d_in[4];  // (h,n,2,dh2)
    const float* values = (const float*)d_in[5];  // (n^2, dim)
    float* out = (float*)d_out;

    char* ws = (char*)d_ws;
    float*    qraw  = (float*)   (ws + 0);              //  8 MB
    _Float16* xb    = (_Float16*)(ws + (8u   << 20));   //  4 MB
    _Float16* wqT   = (_Float16*)(ws + (12u  << 20));   //  2 MB (transposed)
    _Float16* keysh = (_Float16*)(ws + (14u  << 20));   //  1 MB
    _Float16* qh    = (_Float16*)(ws + (15u  << 20));   //  4 MB (post-LN)
    float*    dots  = (float*)   (ws + (19u  << 20));   // 32 MB
    float*    sc    = (float*)   (ws + (51u  << 20));   //  2 MB
    int*      ixr   = (int*)     (ws + (53u  << 20));   //  2 MB
    float*    attn  = (float*)   (ws + (55u  << 20));   //  1 MB
    int*      vidx  = (int*)     (ws + (56u  << 20));   //  1 MB

    k_f32_to_f16      <<<(TB * DIM) / 4 / 256, 256, 0, stream>>>(x, xb);
    k_wq_transpose_f16<<<(DIM * DIM) / 256,    256, 0, stream>>>(Wq, wqT);
    k_f32_to_f16      <<<(HEADS * NKEYS * 2 * DH2) / 4 / 256, 256, 0, stream>>>(keys, keysh);

    k_qproj<<<512, 128, 0, stream>>>(xb, wqT, qraw);
    k_lnorm<<<TB, 256, 0, stream>>>(qraw, gamma, beta, qh);
    k_dots<<<2048, 128, 0, stream>>>(qh, keysh, dots);
    k_top32_512<<<(TB * HEADS * 2) / 4, 128, 0, stream>>>(dots, sc, ixr);
    k_combine<<<(TB * HEADS) / 4, 128, 0, stream>>>(sc, ixr, attn, vidx);
    k_output<<<TB, 256, 0, stream>>>(attn, vidx, values, out);
}
